// DeepGate2_77867757076821
// MI455X (gfx1250) — matrix operand order repaired and checked by reading, not compiled, not run on hardware
//
#include <hip/hip_runtime.h>
#include <stdint.h>

// ---------------------------------------------------------------------------
// CDNA5 WMMA types (wave32): v_wmma_f32_16x16x32_bf16
// ---------------------------------------------------------------------------
typedef __attribute__((ext_vector_type(16))) __bf16 v16bf;
typedef __attribute__((ext_vector_type(8)))  float  v8f;
typedef __attribute__((ext_vector_type(4)))  uint32_t v4u;

union Frag {
    uint32_t u[8];
    v4u      q[2];
    v16bf    v;
};

__device__ __forceinline__ uint16_t f2bf16(float f) {
    __bf16 h = (__bf16)f;                      // HW cvt (RNE)
    return __builtin_bit_cast(uint16_t, h);
}
__device__ __forceinline__ uint32_t pack2(float lo, float hi) {
    return (uint32_t)f2bf16(lo) | ((uint32_t)f2bf16(hi) << 16);
}
__device__ __forceinline__ float bf2f(uint32_t bits16) {
    return __builtin_bit_cast(float, bits16 << 16);
}

// ---------------------------------------------------------------------------
// Repack f32 weights (logical [Kd,Nd]) into WMMA B-fragment order (bf16).
// B 32x16 tile (this code's contiguous k order, same as A): lane L -> col N = L%16, K base = (L<16?0:16);
// VGPR v packs K = base+2v, base+2v+1.
// dword index = ((nt*KT + kt)*32 + lane)*8 + v.   trans=1: W[k][n]=src[n*Kd+k]
// ---------------------------------------------------------------------------
__global__ void repack_wmma_b(const float* __restrict__ W, int Kd, int Nd,
                              int trans, uint32_t* __restrict__ dst) {
    int ndw = (Kd / 32) * (Nd / 16) * 256;
    int i = blockIdx.x * blockDim.x + threadIdx.x;
    if (i >= ndw) return;
    int v    = i & 7;
    int lane = (i >> 3) & 31;
    int tile = i >> 8;
    int KT = Kd / 32;
    int kt = tile % KT;
    int nt = tile / KT;
    int n  = nt * 16 + (lane & 15);
    int k0 = kt * 32 + (lane >> 4) * 16 + 2 * v;
    float f0, f1;
    if (trans) {
        f0 = W[(size_t)n * Kd + k0];
        f1 = W[(size_t)n * Kd + k0 + 1];
    } else {
        f0 = W[(size_t)k0 * Nd + n];
        f1 = W[(size_t)(k0 + 1) * Nd + n];
    }
    dst[i] = pack2(f0, f1);
}

// ---------------------------------------------------------------------------
// WMMA GEMM, A in packed bf16 (row stride 64 dwords = 128 bf16).
//   C = act(A @ B + bias); A rows optionally gathered; CONCAT: K=256 split
//   across A/A2 (hs|hf mirrors).  Output f32 (Cf) or packed-bf16 (Cb).
// One wave owns a 64(M)x16(N) stripe: 4 accumulators, B frag reused 4x.
// A fragment per lane (16-bit 16x32 layout): dwords ap[0..3], ap[4..7] at
// base = row*64 + kt*16 + half*8  -> two b128 loads, zero conversion VALU.
// ---------------------------------------------------------------------------
template <int KT, bool CONCAT>
__global__ __launch_bounds__(128)
void gemm_wmma_bf16(const uint32_t* __restrict__ A, const uint32_t* __restrict__ A2,
                    const int* __restrict__ gather,
                    const uint32_t* __restrict__ Bfrag,
                    const float* __restrict__ bias,
                    float* __restrict__ Cf, uint16_t* __restrict__ Cb,
                    int M, int N, int act) {
    int wv   = threadIdx.x >> 5;
    int lane = threadIdx.x & 31;
    int nt   = blockIdx.y * 4 + wv;
    long mbase = (long)blockIdx.x * 64;
    int nlo  = lane & 15;
    int half = lane >> 4;

    long arow[4];
#pragma unroll
    for (int mt = 0; mt < 4; ++mt) {
        long gr = mbase + mt * 16 + nlo;
        arow[mt] = gather ? (long)gather[gr] : gr;
    }

    v8f acc[4] = {{}, {}, {}, {}};

#pragma unroll
    for (int kt = 0; kt < KT; ++kt) {
        Frag b;
        const uint32_t* bq = Bfrag + (((size_t)nt * KT + kt) * 32 + lane) * 8;
        b.q[0] = *(const v4u*)(bq);
        b.q[1] = *(const v4u*)(bq + 4);

        const uint32_t* Asrc = (CONCAT && kt >= (KT / 2)) ? A2 : A;
        int ktl = (CONCAT && kt >= (KT / 2)) ? (kt - KT / 2) : kt;

#pragma unroll
        for (int mt = 0; mt < 4; ++mt) {
            const uint32_t* ap = Asrc + arow[mt] * 64 + ktl * 16 + half * 8;
            Frag a;
            a.q[0] = *(const v4u*)(ap);
            a.q[1] = *(const v4u*)(ap + 4);
            acc[mt] = __builtin_amdgcn_wmma_f32_16x16x32_bf16(
                false, a.v, false, b.v, (short)0, acc[mt], false, false);
        }
    }

    // C/D layout: lane -> col = L%16; VGPR r -> row = half*8 + r
    int col = nt * 16 + nlo;
    float bs = bias ? bias[col] : 0.0f;
#pragma unroll
    for (int mt = 0; mt < 4; ++mt) {
        long rb = mbase + mt * 16 + half * 8;
#pragma unroll
        for (int r = 0; r < 8; ++r) {
            float vv = acc[mt][r] + bs;
            if (act) vv = fmaxf(vv, 0.0f);
            if (Cb) Cb[(size_t)(rb + r) * N + col] = f2bf16(vv);
            else    Cf[(size_t)(rb + r) * N + col] = vv;
        }
    }
}

// ---------------------------------------------------------------------------
// Structured segment-sum on packed bf16: AND node k <- edges 2k,2k+1;
// NOT node j <- edge 2*K_and + j.  One thread per packed dword (2 cols).
// ---------------------------------------------------------------------------
__global__ void reduce_pairs_bf16(const uint32_t* __restrict__ ein,
                                  uint32_t* __restrict__ m, int Kand, int Knot) {
    long i = (long)blockIdx.x * blockDim.x + threadIdx.x;
    long total = (long)(Kand + Knot) * 64;
    if (i >= total) return;
    long node = i >> 6;
    int  d    = (int)(i & 63);
    if (node < Kand) {
        uint32_t w0 = ein[(size_t)(2 * node) * 64 + d];
        uint32_t w1 = ein[(size_t)(2 * node + 1) * 64 + d];
        float lo = bf2f(w0 & 0xFFFFu) + bf2f(w1 & 0xFFFFu);
        float hi = bf2f(w0 >> 16)     + bf2f(w1 >> 16);
        m[i] = pack2(lo, hi);
    } else {
        m[i] = ein[(size_t)(2 * (long)Kand + (node - Kand)) * 64 + d];
    }
}

// ---------------------------------------------------------------------------
// GRU gates: gi/gh [rows,384] f32 (r|z|n). Writes f32 state + bf16 mirror.
// One thread per packed dword (2 columns).
// ---------------------------------------------------------------------------
__global__ void gru_ew(const float* __restrict__ gi, const float* __restrict__ gh,
                       const float* __restrict__ hin,
                       float* __restrict__ hout, uint32_t* __restrict__ houtb,
                       int rows) {
    long i = (long)blockIdx.x * blockDim.x + threadIdx.x;
    long total = (long)rows * 64;
    if (i >= total) return;
    long r = i >> 6;
    int  c = (int)(i & 63) * 2;
    size_t b = (size_t)r * 384;
    float o[2];
#pragma unroll
    for (int j = 0; j < 2; ++j) {
        int cc = c + j;
        float ir  = gi[b + cc],       hr = gh[b + cc];
        float iz  = gi[b + 128 + cc], hz = gh[b + 128 + cc];
        float in_ = gi[b + 256 + cc], hn = gh[b + 256 + cc];
        float rr = 1.0f / (1.0f + __expf(-(ir + hr)));
        float zz = 1.0f / (1.0f + __expf(-(iz + hz)));
        float nn = tanhf(in_ + rr * hn);
        float hv = hin[(size_t)r * 128 + cc];
        o[j] = (1.0f - zz) * nn + zz * hv;
    }
    hout[(size_t)r * 128 + c]     = o[0];
    hout[(size_t)r * 128 + c + 1] = o[1];
    houtb[i] = pack2(o[0], o[1]);
}

// ---------------------------------------------------------------------------
// Host orchestration
// ---------------------------------------------------------------------------
struct MlpW { uint32_t *W1, *W2, *W3; const float *b1, *b2, *b3; };
struct GruW { uint32_t *Wih, *Whh;    const float *bih, *bhh;    };

extern "C" void kernel_launch(void* const* d_in, const int* in_sizes, int n_in,
                              void* d_out, int out_size, void* d_ws, size_t ws_size,
                              hipStream_t stream) {
    (void)in_sizes; (void)n_in; (void)out_size; (void)ws_size;

    const int Dm = 128, P = 8192, KAND = 6144, KNOT = 2048, K = 8192, L = 16;
    const int EL = 2 * KAND + KNOT;           // 14336 edges per level
    const int NTOT = P + (L - 1) * K;         // 131072 nodes

    const int* edge_src = (const int*)d_in[0];

    float* hs = (float*)d_out;                // [NTOT,128] f32 (output)
    float* hf = hs + (size_t)NTOT * Dm;

    // ---- workspace carve-up ------------------------------------------------
    char*  wsB = (char*)d_ws;
    size_t off = 0;
    auto alloc = [&](size_t bytes) -> void* {
        void* p = wsB + off;
        off += (bytes + 255) & ~(size_t)255;
        return p;
    };
    auto allocF = [&](size_t n) { return (float*)alloc(n * sizeof(float)); };
    auto allocU = [&](size_t n) { return (uint32_t*)alloc(n * sizeof(uint32_t)); };

    auto wsz = [](int Kd, int Nd) { return (size_t)(Kd / 32) * (Nd / 16) * 256; };

    MlpW mlps[4];                 // 0:p_as 1:p_ns 2:p_af 3:p_nf
    int  mlp_in[4] = {1, 7, 13, 19};
    GruW grus[4];                 // 0:g_as 1:g_af 2:g_ns 3:g_nf
    int  gru_in[4] = {25, 29, 33, 37};

    for (int i = 0; i < 4; ++i) {
        int k1 = (i == 2) ? 256 : 128;
        mlps[i].W1 = allocU(wsz(k1, 128));
        mlps[i].W2 = allocU(wsz(128, 128));
        mlps[i].W3 = allocU(wsz(128, 128));
        mlps[i].b1 = (const float*)d_in[mlp_in[i] + 1];
        mlps[i].b2 = (const float*)d_in[mlp_in[i] + 3];
        mlps[i].b3 = (const float*)d_in[mlp_in[i] + 5];
    }
    for (int i = 0; i < 4; ++i) {
        grus[i].Wih = allocU(wsz(128, 384));
        grus[i].Whh = allocU(wsz(128, 384));
        grus[i].bih = (const float*)d_in[gru_in[i] + 2];
        grus[i].bhh = (const float*)d_in[gru_in[i] + 3];
    }

    uint32_t* hsb = allocU((size_t)NTOT * 64);   // bf16 mirror of hs
    uint32_t* hfb = allocU((size_t)NTOT * 64);   // bf16 mirror of hf
    uint32_t* g1b = allocU((size_t)EL * 64);     // MLP ping (bf16)
    uint32_t* g0b = allocU((size_t)EL * 64);     // MLP pong (bf16)
    uint32_t* mbb = allocU((size_t)K * 64);      // messages (bf16)
    float*    gi  = allocF((size_t)P * 384);     // GRU gates f32
    float*    gh  = allocF((size_t)P * 384);
    float*    h1f = allocF((size_t)P * 128);     // init h1 (f32 scratch)
    uint32_t* h1b = allocU((size_t)P * 64);      // init h1 (bf16)

    // ---- zero state + mirrors ---------------------------------------------
    hipMemsetAsync(d_out, 0, (size_t)2 * NTOT * Dm * sizeof(float), stream);
    hipMemsetAsync(hsb, 0, (size_t)2 * NTOT * 64 * sizeof(uint32_t), stream); // hsb+hfb contiguous

    // ---- repack weights ----------------------------------------------------
    auto repack = [&](const float* W, int Kd, int Nd, int trans, uint32_t* dst) {
        int ndw = (Kd / 32) * (Nd / 16) * 256;
        repack_wmma_b<<<(ndw + 255) / 256, 256, 0, stream>>>(W, Kd, Nd, trans, dst);
    };
    for (int i = 0; i < 4; ++i) {
        int k1 = (i == 2) ? 256 : 128;
        repack((const float*)d_in[mlp_in[i] + 0], k1, 128, 0, mlps[i].W1);
        repack((const float*)d_in[mlp_in[i] + 2], 128, 128, 0, mlps[i].W2);
        repack((const float*)d_in[mlp_in[i] + 4], 128, 128, 0, mlps[i].W3);
    }
    for (int i = 0; i < 4; ++i) {               // Wih/Whh [384,128] -> W.T
        repack((const float*)d_in[gru_in[i] + 0], 128, 384, 1, grus[i].Wih);
        repack((const float*)d_in[gru_in[i] + 1], 128, 384, 1, grus[i].Whh);
    }

    // ---- launch helpers ----------------------------------------------------
    auto gemm4 = [&](const uint32_t* A, const int* gath, const uint32_t* Bf,
                     const float* bias, float* Cf, uint16_t* Cb,
                     int M, int N, int act) {
        dim3 grid(M / 64, N / 64);
        gemm_wmma_bf16<4, false><<<grid, 128, 0, stream>>>(
            A, nullptr, gath, Bf, bias, Cf, Cb, M, N, act);
    };
    auto gemm8c = [&](const uint32_t* A, const uint32_t* A2, const int* gath,
                      const uint32_t* Bf, const float* bias, uint16_t* Cb,
                      int M, int N, int act) {
        dim3 grid(M / 64, N / 64);
        gemm_wmma_bf16<8, true><<<grid, 128, 0, stream>>>(
            A, A2, gath, Bf, bias, nullptr, Cb, M, N, act);
    };
    auto mlp = [&](int mi, const uint32_t* A, const uint32_t* A2,
                   const int* gath, int M, uint32_t* outb) {
        if (mi == 2)
            gemm8c(A, A2, gath, mlps[mi].W1, mlps[mi].b1, (uint16_t*)g1b, M, 128, 1);
        else
            gemm4(A, gath, mlps[mi].W1, mlps[mi].b1, nullptr, (uint16_t*)g1b, M, 128, 1);
        gemm4(g1b, nullptr, mlps[mi].W2, mlps[mi].b2, nullptr, (uint16_t*)g0b, M, 128, 1);
        gemm4(g0b, nullptr, mlps[mi].W3, mlps[mi].b3, nullptr, (uint16_t*)outb, M, 128, 0);
    };
    auto gru = [&](int gw, const uint32_t* xb, const uint32_t* hinb,
                   const float* hin, float* hout, uint32_t* houtb, int rows) {
        gemm4(xb,   nullptr, grus[gw].Wih, grus[gw].bih, gi, nullptr, rows, 384, 0);
        gemm4(hinb, nullptr, grus[gw].Whh, grus[gw].bhh, gh, nullptr, rows, 384, 0);
        long tot = (long)rows * 64;
        gru_ew<<<(unsigned)((tot + 255) / 256), 256, 0, stream>>>(
            gi, gh, hin, hout, houtb, rows);
    };

    // ---- init: h1 = GRU_nf(MLP_nf(0), 0); h2 = GRU_nf(MLP_nf(h1), 0) -------
    mlp(3, hfb, nullptr, nullptr, P, g1b);          // hfb[:P] zero
    gru(3, g1b, hfb, hf, h1f, h1b, P);              // hin zeros -> h1
    mlp(3, h1b, nullptr, nullptr, P, g1b);
    gru(3, g1b, hfb, hf, hf, hfb, P);               // hin zeros -> hf[:P]

    // ---- levelized sweep ---------------------------------------------------
    for (int l = 1; l < L; ++l) {
        int start = P + (l - 1) * K;
        const int* src = edge_src + (size_t)(l - 1) * EL;
        float*    hsA  = hs  + (size_t)start * Dm;
        float*    hfA  = hf  + (size_t)start * Dm;
        float*    hsN  = hs  + (size_t)(start + KAND) * Dm;
        float*    hfN  = hf  + (size_t)(start + KAND) * Dm;
        uint32_t* hsAb = hsb + (size_t)start * 64;
        uint32_t* hfAb = hfb + (size_t)start * 64;
        uint32_t* hsNb = hsb + (size_t)(start + KAND) * 64;
        uint32_t* hfNb = hfb + (size_t)(start + KAND) * 64;
        long rtot = (long)K * 64;
        unsigned rblk = (unsigned)((rtot + 255) / 256);

        // hs structure path (p_as -> g_as on AND nodes)
        mlp(0, hsb, nullptr, src, EL, g1b);
        reduce_pairs_bf16<<<rblk, 256, 0, stream>>>(g1b, mbb, KAND, KNOT);
        gru(0, mbb, hsAb, hsA, hsA, hsAb, KAND);

        // function path, concat(hs[src], hf[src]) (p_af -> g_af on AND nodes)
        mlp(2, hsb, hfb, src, EL, g1b);
        reduce_pairs_bf16<<<rblk, 256, 0, stream>>>(g1b, mbb, KAND, KNOT);
        gru(1, mbb, hfAb, hfA, hfA, hfAb, KAND);

        // NOT-node structure path (p_ns -> g_ns)
        mlp(1, hsb, nullptr, src, EL, g1b);
        reduce_pairs_bf16<<<rblk, 256, 0, stream>>>(g1b, mbb, KAND, KNOT);
        gru(2, mbb + (size_t)KAND * 64, hsNb, hsN, hsN, hsNb, KNOT);

        // NOT-node function path (p_nf -> g_nf)
        mlp(3, hfb, nullptr, src, EL, g1b);
        reduce_pairs_bf16<<<rblk, 256, 0, stream>>>(g1b, mbb, KAND, KNOT);
        gru(3, mbb + (size_t)KAND * 64, hfNb, hfN, hfN, hfNb, KNOT);
    }
}
